// TerritorioCoder_76965813944642
// MI455X (gfx1250) — compile-verified, matrix operand order checked by reading, not run on hardware
//
#include <hip/hip_runtime.h>
#include <hip/hip_bf16.h>

// ---------------------------------------------------------------------------
// MI455X (gfx1250) transformer block.
// - All matrix math through V_WMMA_F32_16X16X32_BF16 (f32 accumulate).
// - Tile staging global->LDS via Tensor Data Mover (tensor_load_to_lds,
//   TENSORcnt) with double buffering, overlapped with WMMA compute.
// - Weights pre-cast to bf16 AND pre-transposed once per launch so every
//   GEMM tile is a plain row-major 2D tile (TDM-describable, incl. LDS row
//   padding via D# pad_interval/pad_amount).
// - rel_pos (256MB stream) read with nontemporal loads to protect L2.
// ---------------------------------------------------------------------------

typedef __attribute__((ext_vector_type(16))) __bf16 v16bf;
typedef __attribute__((ext_vector_type(8)))  __bf16 v8bf;
typedef __attribute__((ext_vector_type(8)))  float  v8f;

union FragBF { v16bf f; v8bf h[2]; };

static __device__ inline v16bf load_frag16(const __bf16* p) {
  FragBF t;
  t.h[0] = *(const v8bf*)(p);
  t.h[1] = *(const v8bf*)(p + 8);
  return t.f;
}

static __device__ inline v8f wmma_bf16(v16bf a, v16bf b, v8f c) {
  return __builtin_amdgcn_wmma_f32_16x16x32_bf16(false, a, false, b,
                                                 (short)0, c, false, false);
}

// ---------------------------------------------------------------------------
// Tensor Data Mover: stage a 2D bf16 tile (tile_k elems contiguous per row,
// tile_rows rows, row stride in elements) into LDS with 8-elem (16B) row pad
// so LDS rows land at stride 72 bf16 (= 32 DW data + 4 DW pad).
// ---------------------------------------------------------------------------
#if defined(__AMDGCN__) && __has_builtin(__builtin_amdgcn_tensor_load_to_lds)
#define HAVE_TDM 1
typedef __attribute__((ext_vector_type(4))) unsigned int u32x4;
typedef __attribute__((ext_vector_type(8))) int i32x8;
typedef __attribute__((ext_vector_type(4))) int i32x4;

static __device__ inline void tdm_call(u32x4 g0, i32x8 g1, i32x4 g2, i32x4 g3) {
#if defined(__clang_major__) && __clang_major__ >= 23
  i32x8 z = {0, 0, 0, 0, 0, 0, 0, 0};
  __builtin_amdgcn_tensor_load_to_lds(g0, g1, g2, g3, z, 0);
#else
  __builtin_amdgcn_tensor_load_to_lds(g0, g1, g2, g3, 0);
#endif
}

static __device__ inline void tdm_load_tile_2d(
    unsigned lds_byte_off, const void* gptr,
    unsigned tile_k, unsigned tile_rows, unsigned long long row_stride_elems)
{
  unsigned long long ga = (unsigned long long)(uintptr_t)gptr;
  u32x4 g0;
  g0[0] = 1u;                                  // count=1 valid user descriptor
  g0[1] = lds_byte_off;                        // lds_addr
  g0[2] = (unsigned)ga;                        // global_addr[31:0]
  g0[3] = (unsigned)(ga >> 32) | (2u << 30);   // global_addr[56:32] | type=2
  unsigned td0 = 1u << 30, td1 = 1u << 30;     // huge tensor dims: no OOB clamp
  i32x8 g1;
  g1[0] = (int)((1u << 16)                     // data_size = 2 bytes
              | (1u << 20)                     // pad_enable
              | (4u << 22)                     // pad_interval code 4 = 32 DW
              | (3u << 25));                   // pad_amount  code 3 =  4 DW
  g1[1] = (int)((td0 & 0xFFFFu) << 16);                               // td0 lo
  g1[2] = (int)(((td0 >> 16) & 0xFFFFu) | ((td1 & 0xFFFFu) << 16));   // td0 hi|td1 lo
  g1[3] = (int)(((td1 >> 16) & 0xFFFFu) | ((tile_k & 0xFFFFu) << 16)); // td1 hi|tile_dim0
  g1[4] = (int)(tile_rows & 0xFFFFu);          // tile_dim1 (tile_dim2 = 0)
  g1[5] = (int)(unsigned)(row_stride_elems & 0xFFFFFFFFull);          // stride0 lo
  g1[6] = (int)(unsigned)((row_stride_elems >> 32) & 0xFFFFull);      // stride0 hi
  g1[7] = 0;
  i32x4 gz = {0, 0, 0, 0};
  tdm_call(g0, g1, gz, gz);
}
#endif

// ---------------------------------------------------------------------------
// bf16 GEMM: C(MxN) = A(MxK) * BT(NxK)^T.  A row-major, B pre-transposed.
// Block 128x64, 8 waves (4x2), each wave 32x32 (4 WMMA accumulators), BK=64.
// Both tiles staged by TDM (double-buffered) when available.
// ---------------------------------------------------------------------------
enum { EPI_BF16 = 0, EPI_QKV = 1, EPI_ADDF32 = 2 };

template<int EPI>
__global__ __launch_bounds__(256) void gemm_bf16_wmma(
    const __bf16* __restrict__ A, const __bf16* __restrict__ BT,
    float* __restrict__ Cf, __bf16* __restrict__ Cb,
    const float* __restrict__ Res,
    __bf16* __restrict__ Qo, __bf16* __restrict__ Ko, __bf16* __restrict__ Vo,
    int M, int N, int K)
{
  constexpr int BM = 128, BN = 64, BK = 64, LDA = 72, LDB = 72;
  __shared__ __bf16 As[2][BM * LDA + 16];   // [m][k], padded rows (+pad slack)
  __shared__ __bf16 Bs[2][BN * LDB + 16];   // [n][k], padded rows

  const int tid  = threadIdx.x;
  const int lane = tid & 31, wave = tid >> 5;
  const int lr   = lane & 15, hi  = lane >> 4;
  const int wm   = wave >> 1, wn  = wave & 1;
  const int m0   = blockIdx.y * BM, n0 = blockIdx.x * BN;

  v8f c00 = {}, c01 = {}, c10 = {}, c11 = {};

#if HAVE_TDM
  if (wave == 0) {
    tdm_load_tile_2d((unsigned)(uintptr_t)&As[0][0],
                     A + (size_t)m0 * K, BK, BM, (unsigned long long)K);
    tdm_load_tile_2d((unsigned)(uintptr_t)&Bs[0][0],
                     BT + (size_t)n0 * K, BK, BN, (unsigned long long)K);
  }
  int p = 0;
  for (int k0 = 0; k0 < K; k0 += BK, p ^= 1) {
    if (wave == 0) __builtin_amdgcn_s_wait_tensorcnt(0);
    __syncthreads();                       // tile p visible to all waves
    if (wave == 0 && k0 + BK < K) {        // prefetch next tile into buffer p^1
      tdm_load_tile_2d((unsigned)(uintptr_t)&As[p ^ 1][0],
                       A + (size_t)m0 * K + k0 + BK, BK, BM, (unsigned long long)K);
      tdm_load_tile_2d((unsigned)(uintptr_t)&Bs[p ^ 1][0],
                       BT + (size_t)n0 * K + k0 + BK, BK, BN, (unsigned long long)K);
    }
#pragma unroll
    for (int kk = 0; kk < BK; kk += 32) {
      v16bf a0 = load_frag16(&As[p][(wm * 32 +      lr) * LDA + kk + hi * 16]);
      v16bf a1 = load_frag16(&As[p][(wm * 32 + 16 + lr) * LDA + kk + hi * 16]);
      v16bf b0 = load_frag16(&Bs[p][(wn * 32 +      lr) * LDB + kk + hi * 16]);
      v16bf b1 = load_frag16(&Bs[p][(wn * 32 + 16 + lr) * LDB + kk + hi * 16]);
      c00 = wmma_bf16(a0, b0, c00);
      c01 = wmma_bf16(a0, b1, c01);
      c10 = wmma_bf16(a1, b0, c10);
      c11 = wmma_bf16(a1, b1, c11);
    }
    __syncthreads();                       // all reads of buffer p done
  }
#else
  for (int k0 = 0; k0 < K; k0 += BK) {
#pragma unroll
    for (int i = 0; i < 4; i++) {          // A tile: straight copy
      int ch = tid + i * 256;
      int r = ch >> 3, co = (ch & 7) * 8;
      *(v8bf*)&As[0][r * LDA + co] = *(const v8bf*)&A[(size_t)(m0 + r) * K + k0 + co];
    }
#pragma unroll
    for (int i = 0; i < 2; i++) {          // B tile: straight copy (pre-transposed)
      int ch = tid + i * 256;
      int r = ch >> 3, co = (ch & 7) * 8;
      *(v8bf*)&Bs[0][r * LDB + co] = *(const v8bf*)&BT[(size_t)(n0 + r) * K + k0 + co];
    }
    __syncthreads();
#pragma unroll
    for (int kk = 0; kk < BK; kk += 32) {
      v16bf a0 = load_frag16(&As[0][(wm * 32 +      lr) * LDA + kk + hi * 16]);
      v16bf a1 = load_frag16(&As[0][(wm * 32 + 16 + lr) * LDA + kk + hi * 16]);
      v16bf b0 = load_frag16(&Bs[0][(wn * 32 +      lr) * LDB + kk + hi * 16]);
      v16bf b1 = load_frag16(&Bs[0][(wn * 32 + 16 + lr) * LDB + kk + hi * 16]);
      c00 = wmma_bf16(a0, b0, c00);
      c01 = wmma_bf16(a0, b1, c01);
      c10 = wmma_bf16(a1, b0, c10);
      c11 = wmma_bf16(a1, b1, c11);
    }
    __syncthreads();
  }
#endif

  // Epilogue. C/D f32 layout: v8f elem i -> row = hi*8+i, col = lr.
#pragma unroll
  for (int tm = 0; tm < 2; tm++) {
#pragma unroll
    for (int tn = 0; tn < 2; tn++) {
      v8f acc = (tm == 0) ? ((tn == 0) ? c00 : c01) : ((tn == 0) ? c10 : c11);
      int gcol = n0 + wn * 32 + tn * 16 + lr;
#pragma unroll
      for (int i = 0; i < 8; i++) {
        int grow = m0 + wm * 32 + tm * 16 + hi * 8 + i;
        float val = acc[i];
        if constexpr (EPI == EPI_BF16) {
          Cb[(size_t)grow * N + gcol] = (__bf16)val;
        } else if constexpr (EPI == EPI_ADDF32) {
          size_t idx = (size_t)grow * N + gcol;
          Cf[idx] = val + Res[idx];
        } else {                        // EPI_QKV scatter; V written (B,H,HD,L)
          int part = gcol >> 10;
          int hh = (gcol >> 6) & 15, dd = gcol & 63;
          int bb = grow >> 11,      ll = grow & 2047;
          __bf16 bv = (__bf16)val;
          if (part == 0)
            Qo[(((size_t)(bb * 16 + hh)) * 2048 + ll) * 64 + dd] = bv;
          else if (part == 1)
            Ko[(((size_t)(bb * 16 + hh)) * 2048 + ll) * 64 + dd] = bv;
          else
            Vo[(((size_t)(bb * 16 + hh)) * 64 + dd) * 2048 + ll] = bv;  // transposed
        }
      }
    }
  }
}

// ---------------------------------------------------------------------------
// Flash attention. 64 queries per block, 4 waves. Q/K fragments load straight
// from global (HD contiguous). V is pre-transposed (B,H,HD,L): its tile is a
// plain 2D tile -> staged by TDM, overlapped with S=QK^T WMMA + softmax.
// ---------------------------------------------------------------------------
__global__ __launch_bounds__(128) void attn_wmma(
    const __bf16* __restrict__ Q, const __bf16* __restrict__ Km,
    const __bf16* __restrict__ Vt, const float* __restrict__ rel,
    __bf16* __restrict__ ctx)
{
  constexpr int LDS_S = 68, LDP = 72, LDVT = 72;
  __shared__ float  Ss[64 * LDS_S];
  __shared__ __bf16 Ps[64 * LDP];
  __shared__ __bf16 Vts[64 * LDVT + 16];   // [d][key], padded rows
  __shared__ float  m_run[64], l_run[64], alpha_s[64];

  const int tid  = threadIdx.x;
  const int lane = tid & 31, wave = tid >> 5;
  const int lr   = lane & 15, hi  = lane >> 4;
  const int qt = blockIdx.x, h = blockIdx.y, b = blockIdx.z;
  const int q0 = qt * 64;
  const float scale = 0.125f;

  const __bf16* qh  = Q  + (((size_t)b * 16 + h) * 2048) * 64;
  const __bf16* kh  = Km + (((size_t)b * 16 + h) * 2048) * 64;
  const __bf16* vhT = Vt + (((size_t)b * 16 + h) * 64) * 2048;
  const float* relh = rel + (size_t)h * 2048 * 2048;

  if (tid < 64) { m_run[tid] = -3.0e38f; l_run[tid] = 0.0f; }

  const int qrow_frag = q0 + wave * 16 + lr;
  v16bf qa0 = load_frag16(&qh[(size_t)qrow_frag * 64 +  0 + hi * 16]);
  v16bf qa1 = load_frag16(&qh[(size_t)qrow_frag * 64 + 32 + hi * 16]);

  v8f o0 = {}, o1 = {}, o2 = {}, o3 = {};
  __syncthreads();

  for (int kt = 0; kt <= qt; kt++) {
    const int kbase = kt * 64;
#if HAVE_TDM
    if (wave == 0)                         // async V tile, overlapped w/ S+softmax
      tdm_load_tile_2d((unsigned)(uintptr_t)Vts, vhT + kbase, 64, 64, 2048ull);
#else
#pragma unroll
    for (int i = 0; i < 4; i++) {
      int ch = tid + i * 128;
      int d = ch >> 3, co = (ch & 7) * 8;
      *(v8bf*)&Vts[d * LDVT + co] = *(const v8bf*)&vhT[(size_t)d * 2048 + kbase + co];
    }
#endif
    // S = Q K^T
#pragma unroll
    for (int tn = 0; tn < 4; tn++) {
      v8f s = {};
      int key = kbase + tn * 16 + lr;
      v16bf kb0 = load_frag16(&kh[(size_t)key * 64 +  0 + hi * 16]);
      s = wmma_bf16(qa0, kb0, s);
      v16bf kb1 = load_frag16(&kh[(size_t)key * 64 + 32 + hi * 16]);
      s = wmma_bf16(qa1, kb1, s);
#pragma unroll
      for (int i = 0; i < 8; i++) {
        int rl = wave * 16 + hi * 8 + i;
        int qrow = q0 + rl;
        float bias = __builtin_nontemporal_load(&relh[(size_t)qrow * 2048 + key]);
        float val = s[i] * scale + bias;
        if (key > qrow) val = -3.0e38f;
        Ss[rl * LDS_S + tn * 16 + lr] = val;
      }
    }
    __syncthreads();

    // Online softmax, one thread per query row
    if (tid < 64) {
      int r = tid;
      float mold = m_run[r], mx = mold;
#pragma unroll 8
      for (int j = 0; j < 64; j++) mx = fmaxf(mx, Ss[r * LDS_S + j]);
      float a = __expf(mold - mx);
      float l = l_run[r] * a;
#pragma unroll 8
      for (int j = 0; j < 64; j++) {
        float pv = __expf(Ss[r * LDS_S + j] - mx);
        l += pv;
        Ps[r * LDP + j] = (__bf16)pv;
      }
      m_run[r] = mx; l_run[r] = l; alpha_s[r] = a;
    }
#if HAVE_TDM
    if (wave == 0) __builtin_amdgcn_s_wait_tensorcnt(0);
#endif
    __syncthreads();

    // O = O*alpha + P.V
#pragma unroll
    for (int i = 0; i < 8; i++) {
      float av = alpha_s[wave * 16 + hi * 8 + i];
      o0[i] *= av; o1[i] *= av; o2[i] *= av; o3[i] *= av;
    }
#pragma unroll
    for (int kk2 = 0; kk2 < 2; kk2++) {
      v16bf pa = load_frag16(&Ps[(wave * 16 + lr) * LDP + kk2 * 32 + hi * 16]);
      v16bf v0 = load_frag16(&Vts[( 0 + lr) * LDVT + kk2 * 32 + hi * 16]);
      o0 = wmma_bf16(pa, v0, o0);
      v16bf v1 = load_frag16(&Vts[(16 + lr) * LDVT + kk2 * 32 + hi * 16]);
      o1 = wmma_bf16(pa, v1, o1);
      v16bf v2 = load_frag16(&Vts[(32 + lr) * LDVT + kk2 * 32 + hi * 16]);
      o2 = wmma_bf16(pa, v2, o2);
      v16bf v3 = load_frag16(&Vts[(48 + lr) * LDVT + kk2 * 32 + hi * 16]);
      o3 = wmma_bf16(pa, v3, o3);
    }
    __syncthreads();                       // Vts/Ps free for next iteration
  }

  // Normalize, write ctx (B, L, H*HD) bf16
#pragma unroll
  for (int tn = 0; tn < 4; tn++) {
    v8f o = (tn == 0) ? o0 : ((tn == 1) ? o1 : ((tn == 2) ? o2 : o3));
    int d = tn * 16 + lr;
#pragma unroll
    for (int i = 0; i < 8; i++) {
      int rl = wave * 16 + hi * 8 + i;
      int qrow = q0 + rl;
      float val = o[i] / l_run[rl];
      ctx[(((size_t)b * 2048 + qrow) * 1024) + h * 64 + d] = (__bf16)val;
    }
  }
}

// ---------------------------------------------------------------------------
__global__ __launch_bounds__(256) void ln_bf16(
    const float* __restrict__ X, const float* __restrict__ act,
    const float* __restrict__ gamma, const float* __restrict__ beta,
    float* __restrict__ xw_out, __bf16* __restrict__ Y)
{
  __shared__ float red[256];
  const int t = blockIdx.x;
  const int tid = threadIdx.x;
  const float aw = act ? act[t] : 1.0f;
  const float* xr = X + (size_t)t * 1024;

  float v[4], s = 0.f, s2 = 0.f;
#pragma unroll
  for (int i = 0; i < 4; i++) {
    int c = tid + i * 256;
    float x = xr[c] * aw;
    v[i] = x; s += x; s2 += x * x;
    if (xw_out) xw_out[(size_t)t * 1024 + c] = x;
  }
  red[tid] = s; __syncthreads();
  for (int o = 128; o > 0; o >>= 1) { if (tid < o) red[tid] += red[tid + o]; __syncthreads(); }
  float mean = red[0] * (1.0f / 1024.0f);
  __syncthreads();
  red[tid] = s2; __syncthreads();
  for (int o = 128; o > 0; o >>= 1) { if (tid < o) red[tid] += red[tid + o]; __syncthreads(); }
  float var  = red[0] * (1.0f / 1024.0f) - mean * mean;
  float rstd = rsqrtf(var + 1e-5f);
#pragma unroll
  for (int i = 0; i < 4; i++) {
    int c = tid + i * 256;
    Y[(size_t)t * 1024 + c] = (__bf16)((v[i] - mean) * rstd * gamma[c] + beta[c]);
  }
}

__global__ void swiglu_bf16(const __bf16* __restrict__ P, __bf16* __restrict__ U)
{
  size_t n = (size_t)4096 * 4096;
  size_t stride = (size_t)gridDim.x * blockDim.x;
  for (size_t i = (size_t)blockIdx.x * blockDim.x + threadIdx.x; i < n; i += stride) {
    size_t tk = i >> 12, j = i & 4095;
    float a = (float)P[tk * 8192 + j];
    float g = (float)P[tk * 8192 + 4096 + j];
    float si = a / (1.0f + __expf(-a));
    U[i] = (__bf16)(si * g);
  }
}

// Transposing f32 -> bf16 cast: src (K x N) row-major -> dst (N x K). K = 2^lgK.
__global__ void f32_to_bf16_T(const float* __restrict__ src, __bf16* __restrict__ dst,
                              int N, int lgK, size_t total)
{
  size_t K = (size_t)1 << lgK;
  size_t stride = (size_t)gridDim.x * blockDim.x;
  for (size_t i = (size_t)blockIdx.x * blockDim.x + threadIdx.x; i < total; i += stride) {
    size_t n = i >> lgK, k = i & (K - 1);
    dst[i] = (__bf16)src[k * (size_t)N + n];
  }
}

// wcatT[j][d] = (j < 4096 ? w1 : w3)[d][j]  -> (8192 x 1024) bf16
__global__ void concat_w13T_bf16(const float* __restrict__ w1, const float* __restrict__ w3,
                                 __bf16* __restrict__ wc)
{
  size_t n = (size_t)8192 * 1024;
  size_t stride = (size_t)gridDim.x * blockDim.x;
  for (size_t i = (size_t)blockIdx.x * blockDim.x + threadIdx.x; i < n; i += stride) {
    size_t j = i >> 10, d = i & 1023;
    wc[i] = (__bf16)(j < 4096 ? w1[d * 4096 + j] : w3[d * 4096 + (j - 4096)]);
  }
}

__global__ __launch_bounds__(256) void conf_kernel(
    const float* __restrict__ out, const float* __restrict__ we,
    const float* __restrict__ be, float* __restrict__ conf)
{
  __shared__ float red[256];
  const int b = blockIdx.x;
  const size_t n = (size_t)2048 * 1024;
  const float* ob = out + (size_t)b * n;
  float s = 0.f;
  for (size_t i = threadIdx.x; i < n; i += 256) s += ob[i] * we[i & 1023];
  red[threadIdx.x] = s; __syncthreads();
  for (int o = 128; o > 0; o >>= 1) { if ((int)threadIdx.x < o) red[threadIdx.x] += red[threadIdx.x + o]; __syncthreads(); }
  if (threadIdx.x == 0) {
    float z = red[0] * (1.0f / 2048.0f) + be[0];
    conf[b] = 1.0f / (1.0f + __expf(-z));
  }
}

// ---------------------------------------------------------------------------
extern "C" void kernel_launch(void* const* d_in, const int* in_sizes, int n_in,
                              void* d_out, int out_size, void* d_ws, size_t ws_size,
                              hipStream_t stream)
{
  const float* x    = (const float*)d_in[0];
  const float* act  = (const float*)d_in[1];
  // d_in[2] = mask (unused: causal mask applied analytically)
  const float* ln1g = (const float*)d_in[3];
  const float* ln1b = (const float*)d_in[4];
  const float* Wqkv = (const float*)d_in[5];
  const float* Wout = (const float*)d_in[6];
  const float* rel  = (const float*)d_in[7];
  const float* ln2g = (const float*)d_in[8];
  const float* ln2b = (const float*)d_in[9];
  const float* w1   = (const float*)d_in[10];
  const float* w3   = (const float*)d_in[11];
  const float* w2   = (const float*)d_in[12];
  const float* we   = (const float*)d_in[13];
  const float* be   = (const float*)d_in[14];

  char* ws = (char*)d_ws;
  size_t off = 0;
  auto carve = [&](size_t bytes) -> void* {
    void* p = ws + off; off += (bytes + 255) & ~(size_t)255; return p;
  };
  __bf16* h_bf    = (__bf16*)carve((size_t)4096 * 1024 * 2);
  float*  xw_f    = (float*) carve((size_t)4096 * 1024 * 4);
  __bf16* q_bf    = (__bf16*)carve((size_t)4096 * 1024 * 2);
  __bf16* k_bf    = (__bf16*)carve((size_t)4096 * 1024 * 2);
  __bf16* vT_bf   = (__bf16*)carve((size_t)4096 * 1024 * 2);   // (B,H,HD,L)
  __bf16* ctx_bf  = (__bf16*)carve((size_t)4096 * 1024 * 2);
  float*  h2_f    = (float*) carve((size_t)4096 * 1024 * 4);
  __bf16* g_bf    = (__bf16*)carve((size_t)4096 * 1024 * 2);
  __bf16* pcat_bf = (__bf16*)carve((size_t)4096 * 8192 * 2);
  __bf16* u_bf    = (__bf16*)carve((size_t)4096 * 4096 * 2);
  __bf16* wqkvT   = (__bf16*)carve((size_t)3072 * 1024 * 2);   // (3D x D)
  __bf16* woutT   = (__bf16*)carve((size_t)1024 * 1024 * 2);   // (D x D)
  __bf16* wcatT   = (__bf16*)carve((size_t)8192 * 1024 * 2);   // (2FF x D)
  __bf16* w2T     = (__bf16*)carve((size_t)1024 * 4096 * 2);   // (D x FF)

  // One-time (per launch) weight cast + transpose -> TDM-friendly (N x K) bf16
  f32_to_bf16_T<<<1024, 256, 0, stream>>>(Wqkv, wqkvT, 3072, 10, (size_t)3072 * 1024);
  f32_to_bf16_T<<<512,  256, 0, stream>>>(Wout, woutT, 1024, 10, (size_t)1024 * 1024);
  f32_to_bf16_T<<<1024, 256, 0, stream>>>(w2,   w2T,   1024, 12, (size_t)1024 * 4096);
  concat_w13T_bf16<<<1024, 256, 0, stream>>>(w1, w3, wcatT);

  // xw = x*act ; h = LN1(xw)
  ln_bf16<<<4096, 256, 0, stream>>>(x, act, ln1g, ln1b, xw_f, h_bf);

  // qkv = h @ Wqkv  (scatter -> q,k (B,H,L,HD), v (B,H,HD,L))
  gemm_bf16_wmma<EPI_QKV><<<dim3(3072 / 64, 4096 / 128), 256, 0, stream>>>(
      h_bf, wqkvT, nullptr, nullptr, nullptr, q_bf, k_bf, vT_bf, 4096, 3072, 1024);

  // flash attention (rel_pos bias, causal)
  attn_wmma<<<dim3(2048 / 64, 16, 2), 128, 0, stream>>>(q_bf, k_bf, vT_bf, rel, ctx_bf);

  // h2 = xw + ctx @ Wout
  gemm_bf16_wmma<EPI_ADDF32><<<dim3(1024 / 64, 4096 / 128), 256, 0, stream>>>(
      ctx_bf, woutT, h2_f, nullptr, xw_f, nullptr, nullptr, nullptr, 4096, 1024, 1024);

  // g = LN2(h2)
  ln_bf16<<<4096, 256, 0, stream>>>(h2_f, nullptr, ln2g, ln2b, nullptr, g_bf);

  // pcat = g @ [w1 | w3]
  gemm_bf16_wmma<EPI_BF16><<<dim3(8192 / 64, 4096 / 128), 256, 0, stream>>>(
      g_bf, wcatT, nullptr, pcat_bf, nullptr, nullptr, nullptr, nullptr, 4096, 8192, 1024);

  // u = silu(p1) * p3
  swiglu_bf16<<<2048, 256, 0, stream>>>(pcat_bf, u_bf);

  // out = h2 + u @ w2  -> d_out (f32)
  gemm_bf16_wmma<EPI_ADDF32><<<dim3(1024 / 64, 4096 / 128), 256, 0, stream>>>(
      u_bf, w2T, (float*)d_out, nullptr, h2_f, nullptr, nullptr, nullptr, 4096, 1024, 4096);

  // confianza
  conf_kernel<<<2, 256, 0, stream>>>((const float*)d_out, we, be,
                                     (float*)d_out + (size_t)4096 * 1024);
}
